// MixedMoEProjectionLayer_27290222199136
// MI455X (gfx1250) — compile-verified, matrix operand order checked by reading, not run on hardware
//
#include <hip/hip_runtime.h>
#include <hip/hip_bf16.h>

// MI455X / gfx1250 MoE forward.
// Dense per-expert bf16 WMMA GEMM pipeline (v_wmma_f32_16x16x32_bf16, fp32
// accumulate) with CDNA5 async global->LDS tile DMA (global_load_async_to_lds,
// ASYNCcnt, double-buffered LDS, one barrier per K-step).
// LayerNorm between layers; gate uses pre-renormalization softmax weights
// (matches reference). Workspace requirement ~663 MB.

typedef __bf16 bf16;
typedef bf16  bf16x8   __attribute__((ext_vector_type(8)));
typedef bf16  bf16x16  __attribute__((ext_vector_type(16)));
typedef float f32x8    __attribute__((ext_vector_type(8)));

#define BATCH_N   8192
#define IN_DIM    1024
#define TM        128
#define TN        128
#define TK        32
#define LDS_STR   40   // padded row stride in halves (80B, 16B aligned)

// ---------------------------------------------------------------- activations
template<int ACT>
__device__ __forceinline__ float apply_act(float v) {
  if (ACT == 0) return 0.5f * v * (1.0f + erff(v * 0.70710678118654752f)); // exact GELU
  if (ACT == 1) return v / (1.0f + __expf(-v));                            // SiLU
  if (ACT == 2) return fmaxf(v, 0.0f);                                     // ReLU
  if (ACT == 3) return v > 0.0f ? v : 0.01f * v;                           // LeakyReLU
  return v;                                                                // none
}

// ------------------------------------------------------------- utility kernels
__global__ __launch_bounds__(256) void zero_f32_kernel(float* __restrict__ p, int n) {
  int i = blockIdx.x * 256 + threadIdx.x;
  if (i < n) p[i] = 0.0f;
}

__global__ __launch_bounds__(256) void cvt_f32_to_bf16_kernel(
    const float* __restrict__ src, bf16* __restrict__ dst, int n) {
  int i = blockIdx.x * 256 + threadIdx.x;
  if (i < n) dst[i] = (bf16)src[i];
}

// W [K][N] fp32 (row-major, fan_in x fan_out)  ->  Wt [N][K] bf16
__global__ __launch_bounds__(256) void transpose_cvt_kernel(
    const float* __restrict__ W, bf16* __restrict__ Wt, int K, int N) {
  __shared__ float tile[32][33];
  const int nb = blockIdx.x * 32;
  const int kb = blockIdx.y * 32;
  const int tx = threadIdx.x;        // 0..31
  const int ty = threadIdx.y;        // 0..7
#pragma unroll
  for (int i = 0; i < 32; i += 8)
    tile[ty + i][tx] = W[(size_t)(kb + ty + i) * N + (nb + tx)];
  __syncthreads();
#pragma unroll
  for (int i = 0; i < 32; i += 8)
    Wt[(size_t)(nb + ty + i) * K + (kb + tx)] = (bf16)tile[tx][ty + i];
}

// ------------------------------------------------------------------ gate
// scores = softmax(x @ gw + gb); keep top-2 (pre-renormalization values).
__global__ __launch_bounds__(128) void gate_kernel(
    const float* __restrict__ x, const float* __restrict__ gw,
    const float* __restrict__ gb, float* __restrict__ gate) {
  const int lane = threadIdx.x & 31;
  const int row  = blockIdx.x * 4 + (threadIdx.x >> 5);
  const float* xr = x + (size_t)row * IN_DIM;
  float acc[4] = {0.f, 0.f, 0.f, 0.f};
  for (int i = lane; i < IN_DIM; i += 32) {
    float xv = xr[i];
#pragma unroll
    for (int e = 0; e < 4; ++e) acc[e] += xv * gw[i * 4 + e];
  }
#pragma unroll
  for (int off = 16; off > 0; off >>= 1)
#pragma unroll
    for (int e = 0; e < 4; ++e) acc[e] += __shfl_xor(acc[e], off, 32);
  if (lane == 0) {
    float lo[4], sc[4];
#pragma unroll
    for (int e = 0; e < 4; ++e) lo[e] = acc[e] + gb[e];
    float m = fmaxf(fmaxf(lo[0], lo[1]), fmaxf(lo[2], lo[3]));
    float s = 0.f;
#pragma unroll
    for (int e = 0; e < 4; ++e) { sc[e] = __expf(lo[e] - m); s += sc[e]; }
    float inv = 1.0f / s;
#pragma unroll
    for (int e = 0; e < 4; ++e) sc[e] *= inv;
    int i1 = 0;
#pragma unroll
    for (int e = 1; e < 4; ++e) if (sc[e] > sc[i1]) i1 = e;
    int i2 = -1;
#pragma unroll
    for (int e = 0; e < 4; ++e)
      if (e != i1 && (i2 < 0 || sc[e] > sc[i2])) i2 = e;
#pragma unroll
    for (int e = 0; e < 4; ++e)
      gate[(size_t)row * 4 + e] = (e == i1 || e == i2) ? sc[e] : 0.0f;
  }
}

// ----------------------------------------------------------- async tile DMA
// CDNA5 GLOBAL_LOAD_ASYNC_TO_LDS_B128: copies global memory into LDS,
// 16B per lane, tracked by ASYNCcnt. INST_OFFSET applies to both addresses,
// so one (lds,global) pair covers this thread's 32B (two chunks at +0,+16).
__device__ __forceinline__ void async_copy32B(unsigned int lds_addr,
                                              const bf16* gsrc) {
  asm volatile(
      "global_load_async_to_lds_b128 %0, %1, off\n\t"
      "global_load_async_to_lds_b128 %0, %1, off offset:16"
      :: "v"(lds_addr), "v"(gsrc) : "memory");
}

__device__ __forceinline__ void wait_async_all() {
  asm volatile("s_wait_asynccnt 0x0" ::: "memory");
}

// ------------------------------------------------------------------ WMMA GEMM
// C[M][N] = act(A[M][K](bf16) * Bt[N][K](bf16)^T + bias)   (fp32 out)
template<int ACT>
__global__ __launch_bounds__(256) void gemm_bias_act_kernel(
    const bf16* __restrict__ A, const bf16* __restrict__ Bt,
    const float* __restrict__ bias, float* __restrict__ C,
    int M, int N, int K) {
  __shared__ __attribute__((aligned(16))) bf16 smA[2][TM * LDS_STR];
  __shared__ __attribute__((aligned(16))) bf16 smB[2][TN * LDS_STR];

  const int tid  = threadIdx.x;
  const int lane = tid & 31;
  const int wave = tid >> 5;
  const int wm   = wave & 3;          // 4 waves over M
  const int wn   = wave >> 2;         // 2 waves over N
  const int r16  = lane & 15;
  const int kb   = (lane >> 4) << 3;  // 0 or 8 halves (ISA bf16 A/B layout)

  const int m0 = blockIdx.y * TM;
  const int n0 = blockIdx.x * TN;

  // cooperative tile fill: 128 rows x 32 halves; 2 threads/row, 16 halves each
  const int ldRow = tid >> 1;
  const int ldSeg = (tid & 1) << 4;   // 0 or 16 halves
  const bf16* gA = A  + (size_t)(m0 + ldRow) * K + ldSeg;
  const bf16* gB = Bt + (size_t)(n0 + ldRow) * K + ldSeg;

  // LDS group-segment byte offsets (flat LDS addr[31:0] == LDS offset)
  unsigned int ldsA[2], ldsB[2];
#pragma unroll
  for (int p = 0; p < 2; ++p) {
    ldsA[p] = (unsigned int)(size_t)&smA[p][ldRow * LDS_STR + ldSeg];
    ldsB[p] = (unsigned int)(size_t)&smB[p][ldRow * LDS_STR + ldSeg];
  }

  f32x8 acc[2][4] = {};

  const int nK = K / TK;
  // prologue: async-fill buffer 0 with K-step 0
  async_copy32B(ldsA[0], gA);
  async_copy32B(ldsB[0], gB);

  for (int ks = 0; ks < nK; ++ks) {
    const int par = ks & 1;
    wait_async_all();        // this wave's fill of buf[par] complete
    __syncthreads();         // all waves' fills complete / prior reads done
    if (ks + 1 < nK) {       // DMA next tile under the WMMA work
      async_copy32B(ldsA[par ^ 1], gA + (size_t)(ks + 1) * TK);
      async_copy32B(ldsB[par ^ 1], gB + (size_t)(ks + 1) * TK);
    }
    union { bf16x16 v; bf16x8 h[2]; } af[2], bfr[4];
#pragma unroll
    for (int mi = 0; mi < 2; ++mi) {
      const int base = (wm * 32 + mi * 16 + r16) * LDS_STR + kb;
      af[mi].h[0] = *(const bf16x8*)&smA[par][base];
      af[mi].h[1] = *(const bf16x8*)&smA[par][base + 16];
    }
#pragma unroll
    for (int ni = 0; ni < 4; ++ni) {
      const int base = (wn * 64 + ni * 16 + r16) * LDS_STR + kb;
      bfr[ni].h[0] = *(const bf16x8*)&smB[par][base];
      bfr[ni].h[1] = *(const bf16x8*)&smB[par][base + 16];
    }
#pragma unroll
    for (int mi = 0; mi < 2; ++mi)
#pragma unroll
      for (int ni = 0; ni < 4; ++ni)
        acc[mi][ni] = __builtin_amdgcn_wmma_f32_16x16x32_bf16(
            false, af[mi].v, false, bfr[ni].v, (short)0, acc[mi][ni],
            false, false);
  }

  // epilogue: C/D layout — VGPR j holds row j (lanes 0-15) / j+8 (lanes 16-31)
  const int cn = n0 + wn * 64 + (lane & 15);
  const int rb = m0 + wm * 32 + ((lane >> 4) << 3);
#pragma unroll
  for (int mi = 0; mi < 2; ++mi)
#pragma unroll
    for (int ni = 0; ni < 4; ++ni) {
      const int c  = cn + ni * 16;
      const float bv = bias[c];
#pragma unroll
      for (int j = 0; j < 8; ++j) {
        float v = acc[mi][ni][j] + bv;
        C[(size_t)(rb + mi * 16 + j) * N + c] = apply_act<ACT>(v);
      }
    }
}

// --------------------------------------------------------------- LayerNorm
__global__ __launch_bounds__(256) void ln_to_bf16_kernel(
    const float* __restrict__ H, const float* __restrict__ g,
    const float* __restrict__ beta, bf16* __restrict__ out, int N) {
  const int row = blockIdx.x;
  const float* h = H + (size_t)row * N;
  float s = 0.f, s2 = 0.f;
  for (int i = threadIdx.x; i < N; i += 256) { float v = h[i]; s += v; s2 += v * v; }
  __shared__ float rs[256], rq[256];
  rs[threadIdx.x] = s; rq[threadIdx.x] = s2;
  __syncthreads();
  for (int off = 128; off > 0; off >>= 1) {
    if (threadIdx.x < off) {
      rs[threadIdx.x] += rs[threadIdx.x + off];
      rq[threadIdx.x] += rq[threadIdx.x + off];
    }
    __syncthreads();
  }
  const float mean = rs[0] / N;
  const float var  = rq[0] / N - mean * mean;
  const float rinv = rsqrtf(var + 1e-5f);
  bf16* o = out + (size_t)row * N;
  for (int i = threadIdx.x; i < N; i += 256)
    o[i] = (bf16)((h[i] - mean) * rinv * g[i] + beta[i]);
}

// final: out += gate[row][e] * LayerNorm(H)
__global__ __launch_bounds__(256) void final_ln_accum_kernel(
    const float* __restrict__ H, const float* __restrict__ g,
    const float* __restrict__ beta, const float* __restrict__ gate,
    int e, float* __restrict__ out) {
  const int row = blockIdx.x;
  const int N = IN_DIM;
  const float* h = H + (size_t)row * N;
  float s = 0.f, s2 = 0.f;
  for (int i = threadIdx.x; i < N; i += 256) { float v = h[i]; s += v; s2 += v * v; }
  __shared__ float rs[256], rq[256];
  rs[threadIdx.x] = s; rq[threadIdx.x] = s2;
  __syncthreads();
  for (int off = 128; off > 0; off >>= 1) {
    if (threadIdx.x < off) {
      rs[threadIdx.x] += rs[threadIdx.x + off];
      rq[threadIdx.x] += rq[threadIdx.x + off];
    }
    __syncthreads();
  }
  const float mean = rs[0] / N;
  const float var  = rq[0] / N - mean * mean;
  const float rinv = rsqrtf(var + 1e-5f);
  const float w = gate[(size_t)row * 4 + e];
  float* o = out + (size_t)row * N;
  for (int i = threadIdx.x; i < N; i += 256)
    o[i] += w * ((h[i] - mean) * rinv * g[i] + beta[i]);
}

// ---------------------------------------------------------------- dispatch
static inline void launch_gemm(int act, const bf16* A, const bf16* Bt,
                               const float* bias, float* C,
                               int M, int N, int K, hipStream_t s) {
  dim3 grid(N / TN, M / TM), block(256);
  switch (act) {
    case 0: gemm_bias_act_kernel<0><<<grid, block, 0, s>>>(A, Bt, bias, C, M, N, K); break;
    case 1: gemm_bias_act_kernel<1><<<grid, block, 0, s>>>(A, Bt, bias, C, M, N, K); break;
    case 2: gemm_bias_act_kernel<2><<<grid, block, 0, s>>>(A, Bt, bias, C, M, N, K); break;
    case 3: gemm_bias_act_kernel<3><<<grid, block, 0, s>>>(A, Bt, bias, C, M, N, K); break;
    default: gemm_bias_act_kernel<-1><<<grid, block, 0, s>>>(A, Bt, bias, C, M, N, K); break;
  }
}

extern "C" void kernel_launch(void* const* d_in, const int* in_sizes, int n_in,
                              void* d_out, int out_size, void* d_ws, size_t ws_size,
                              hipStream_t stream) {
  const float* x  = (const float*)d_in[0];
  const float* gw = (const float*)d_in[1];
  const float* gb = (const float*)d_in[2];

  static const int depths[4] = {1, 2, 3, 1};
  static const int hids[4]   = {2048, 4096, 6144, 2048};

  const float *ew[4][3], *eb[4][3], *eg[4][3], *ebt[4][3];
  const float *wo[4], *bo[4], *go[4], *beo[4];
  int p = 3;
  for (int e = 0; e < 4; ++e) {
    for (int l = 0; l < depths[e]; ++l) {
      ew[e][l]  = (const float*)d_in[p++];
      eb[e][l]  = (const float*)d_in[p++];
      eg[e][l]  = (const float*)d_in[p++];
      ebt[e][l] = (const float*)d_in[p++];
    }
    wo[e]  = (const float*)d_in[p++];
    bo[e]  = (const float*)d_in[p++];
    go[e]  = (const float*)d_in[p++];
    beo[e] = (const float*)d_in[p++];
  }

  // ---- workspace layout (bytes) -- needs ~663 MB
  char* ws = (char*)d_ws;
  const size_t B = BATCH_N;
  float* gate_ws = (float*)ws;                               // B*4*4     = 128 KB
  size_t off = 131072;
  bf16* x_bf = (bf16*)(ws + off);   off += B * IN_DIM * 2;   // 16 MB
  bf16* act0 = (bf16*)(ws + off);   off += B * 6144 * 2;     // 96 MB
  bf16* act1 = (bf16*)(ws + off);   off += B * 6144 * 2;     // 96 MB
  float* Cbuf = (float*)(ws + off); off += B * 6144 * 4;     // 192 MB
  bf16* wt_cur = (bf16*)(ws + off);                          // ~232 MB bf16 weights

  // ---- prologue: zero out, convert x, transpose+convert all weights
  {
    int n = (int)(B * IN_DIM);
    zero_f32_kernel<<<n / 256, 256, 0, stream>>>((float*)d_out, n);
    cvt_f32_to_bf16_kernel<<<n / 256, 256, 0, stream>>>(x, x_bf, n);
  }
  bf16* wt_h[4][3];
  bf16* wt_o[4];
  for (int e = 0; e < 4; ++e) {
    int in_dim = IN_DIM;
    for (int l = 0; l < depths[e]; ++l) {
      int N = hids[e];
      wt_h[e][l] = wt_cur;
      transpose_cvt_kernel<<<dim3(N / 32, in_dim / 32), dim3(32, 8), 0, stream>>>(
          ew[e][l], wt_cur, in_dim, N);
      wt_cur += (size_t)in_dim * N;
      in_dim = N;
    }
    wt_o[e] = wt_cur;
    transpose_cvt_kernel<<<dim3(IN_DIM / 32, in_dim / 32), dim3(32, 8), 0, stream>>>(
        wo[e], wt_cur, in_dim, IN_DIM);
    wt_cur += (size_t)in_dim * IN_DIM;
  }

  // ---- gate
  gate_kernel<<<(int)(B / 4), 128, 0, stream>>>(x, gw, gb, gate_ws);

  // ---- experts (dense over full batch; gate weight applied at the end)
  for (int e = 0; e < 4; ++e) {
    const bf16* Ain = x_bf;
    int K = IN_DIM;
    for (int l = 0; l < depths[e]; ++l) {
      const int N = hids[e];
      launch_gemm(e % 4, Ain, wt_h[e][l], eb[e][l], Cbuf, (int)B, N, K, stream);
      bf16* nxt = (l & 1) ? act1 : act0;
      ln_to_bf16_kernel<<<(int)B, 256, 0, stream>>>(Cbuf, eg[e][l], ebt[e][l], nxt, N);
      Ain = nxt;
      K = N;
    }
    launch_gemm(-1, Ain, wt_o[e], bo[e], Cbuf, (int)B, IN_DIM, K, stream);
    final_ln_accum_kernel<<<(int)B, 256, 0, stream>>>(
        Cbuf, go[e], beo[e], gate_ws, e, (float*)d_out);
  }
}
